// ChessNet_63617055589129
// MI455X (gfx1250) — matrix operand order repaired and checked by reading, not compiled, not run on hardware
//
#include <hip/hip_runtime.h>
#include <hip/hip_bf16.h>
#include <math.h>

// ---------------- model dims ----------------
#define BB   128
#define CC   112
#define DD   1024
#define HH   16
#define HDD  64
#define LL   12
#define FFF  2048
#define HCC  32
#define SGHH 256
#define GENN 256
#define NEGV (-1.0e9f)
#define TOK  (BB*64)            // 8192 tokens

typedef __attribute__((ext_vector_type(16))) __bf16 v16bf;
typedef __attribute__((ext_vector_type(8)))  float  v8f;

union FragU { unsigned u[8]; v16bf v; };

// round-to-nearest-even f32 -> bf16
static __device__ inline __bf16 f2bf(float f) {
  unsigned u = __float_as_uint(f);
  unsigned r = (u + 0x7fffu + ((u >> 16) & 1u)) >> 16;
  unsigned short s = (unsigned short)r;
  __bf16 b;
  __builtin_memcpy(&b, &s, 2);
  return b;
}
// pack two f32 into one dword of 2 bf16 (RNE)
static __device__ inline unsigned pack2bf(float a, float b) {
  unsigned ua = __float_as_uint(a); unsigned ra = (ua + 0x7fffu + ((ua >> 16) & 1u)) >> 16;
  unsigned ub = __float_as_uint(b); unsigned rb = (ub + 0x7fffu + ((ub >> 16) & 1u)) >> 16;
  return (ra & 0xffffu) | (rb << 16);
}

static __device__ inline float softplusf(float x) { return (x > 20.f) ? x : log1pf(expf(x)); }
static __device__ inline float mishf(float x)     { return x * tanhf(softplusf(x)); }
static __device__ inline float siluf(float x)     { return x / (1.f + expf(-x)); }

__constant__ int c_dr[8]  = { 1, 1, 0,-1,-1,-1, 0, 1};
__constant__ int c_dc[8]  = { 0, 1, 1, 1, 0,-1,-1,-1};
__constant__ int c_knr[8] = { 2, 1,-1,-2,-2,-1, 1, 2};
__constant__ int c_knc[8] = { 1, 2, 2, 1,-1,-2,-2,-1};

// ============================================================
// Tiled WMMA GEMM:  C[M,N] = epi(A[M,K] @ W[K,N])
// Block tile 128x64, BK=32, 8 waves, 32x32 per wave (4 WMMA/chunk).
// Requires K % 4 == 0 and N % 4 == 0 (true for all uses here).
// EPI: 0 none, 1 +bias, 2 mish(+bias), 3 silu(+bias), 4 +bias+res
// ============================================================
template<int EPI>
__global__ void __launch_bounds__(256)
k_gemm(const float* __restrict__ A, const float* __restrict__ Wm,
       const float* __restrict__ bias, const float* __restrict__ res,
       float* __restrict__ C, int M, int N, int K)
{
  __shared__ __bf16 sA[128][34];  // [m][k], row = 68B (4B aligned, even bf16)
  __shared__ __bf16 sB[64][34];   // [n][k] (W transposed)
  const int tid  = threadIdx.x;
  const int m0   = blockIdx.y * 128, n0 = blockIdx.x * 64;
  const int wave = tid >> 5, lane = tid & 31;
  const int half = lane >> 4, l16 = lane & 15;
  const int wm   = (wave >> 1) * 32, wn = (wave & 1) * 32;
  v8f acc00 = {}; v8f acc01 = {}; v8f acc10 = {}; v8f acc11 = {};

  for (int k0 = 0; k0 < K; k0 += 32) {
    // ---- stage A (128x32) as float4 loads -> packed bf16 dword stores ----
#pragma unroll
    for (int t = 0; t < 4; t++) {
      int q = tid + t * 256;                 // 1024 quads
      int r = q >> 3, c = (q & 7) * 4;
      int m = m0 + r, kk = k0 + c;
      float4 v = {0.f, 0.f, 0.f, 0.f};
      if (m < M && kk < K) v = *(const float4*)&A[(size_t)m * K + kk];
      unsigned* d = (unsigned*)&sA[r][c];
      d[0] = pack2bf(v.x, v.y);
      d[1] = pack2bf(v.z, v.w);
    }
    // ---- stage W chunk (32xN64) transposed into sB[n][k] ----
#pragma unroll
    for (int t = 0; t < 2; t++) {
      int q = tid + t * 256;                 // 512 quads
      int kr = q >> 4, c4 = (q & 15) * 4;
      int kk = k0 + kr, n = n0 + c4;
      float4 v = {0.f, 0.f, 0.f, 0.f};
      if (kk < K && n < N) v = *(const float4*)&Wm[(size_t)kk * N + n];
      sB[c4 + 0][kr] = f2bf(v.x);
      sB[c4 + 1][kr] = f2bf(v.y);
      sB[c4 + 2][kr] = f2bf(v.z);
      sB[c4 + 3][kr] = f2bf(v.w);
    }
    // prefetch next K-chunk while this one is consumed
    if (k0 + 32 < K) {
      int pr = tid & 127;
      if (m0 + pr < M) __builtin_prefetch(&A[(size_t)(m0 + pr) * K + k0 + 32], 0, 0);
      int pk2 = tid & 31;
      if (k0 + 32 + pk2 < K) __builtin_prefetch(&Wm[(size_t)(k0 + 32 + pk2) * N + n0], 0, 0);
    }
    __syncthreads();

    FragU a0, a1, b0, b1;
    const unsigned* pA0 = (const unsigned*)&sA[wm + l16][0];
    const unsigned* pA1 = (const unsigned*)&sA[wm + 16 + l16][0];
    const unsigned* pB0 = (const unsigned*)&sB[wn + l16][0];
    const unsigned* pB1 = (const unsigned*)&sB[wn + 16 + l16][0];
#pragma unroll
    for (int j = 0; j < 4; j++) { a0.u[j] = pA0[j + 4 * half]; a1.u[j] = pA1[j + 4 * half]; }
#pragma unroll
    for (int j = 4; j < 8; j++) { a0.u[j] = pA0[8 + (j - 4) + 4 * half]; a1.u[j] = pA1[8 + (j - 4) + 4 * half]; }
#pragma unroll
    for (int j = 0; j < 8; j++) { b0.u[j] = pB0[(j & 3) + 4 * half + 8 * (j >> 2)]; b1.u[j] = pB1[(j & 3) + 4 * half + 8 * (j >> 2)]; }
    acc00 = __builtin_amdgcn_wmma_f32_16x16x32_bf16(false, a0.v, false, b0.v, (short)0, acc00, false, false);
    acc01 = __builtin_amdgcn_wmma_f32_16x16x32_bf16(false, a0.v, false, b1.v, (short)0, acc01, false, false);
    acc10 = __builtin_amdgcn_wmma_f32_16x16x32_bf16(false, a1.v, false, b0.v, (short)0, acc10, false, false);
    acc11 = __builtin_amdgcn_wmma_f32_16x16x32_bf16(false, a1.v, false, b1.v, (short)0, acc11, false, false);
    __syncthreads();
  }

#pragma unroll
  for (int r = 0; r < 8; r++) {
#pragma unroll
    for (int mt = 0; mt < 2; mt++) {
      int m = m0 + wm + mt * 16 + r + 8 * half;
      if (m >= M) continue;
#pragma unroll
      for (int nt = 0; nt < 2; nt++) {
        int n = n0 + wn + nt * 16 + l16;
        if (n >= N) continue;
        float v0 = (mt == 0) ? ((nt == 0) ? acc00[r] : acc01[r])
                             : ((nt == 0) ? acc10[r] : acc11[r]);
        if (EPI >= 1) v0 += bias[n];
        if (EPI == 2) v0 = mishf(v0);
        if (EPI == 3) v0 = siluf(v0);
        if (EPI == 4) v0 += res[(size_t)m * N + n];
        C[(size_t)m * N + n] = v0;
      }
    }
  }
}

// ============================================================
// Fused attention per (b,h): logits = QK^T*0.125 + bias, softmax, P@V
// ============================================================
__global__ void __launch_bounds__(256)
k_attn(const float* __restrict__ q, const float* __restrict__ k,
       const float* __restrict__ v, const float* __restrict__ bias,
       float* __restrict__ o)
{
  __shared__ __bf16 sQ[64][66];   // Q rows [i][d]; reused for P bf16
  __shared__ __bf16 sK[64][66];   // K rows [j][d]  (== B^T layout for QK^T)
  __shared__ __bf16 sVT[64][66];  // V transposed [d][j]
  __shared__ float  sS[64][65];
  const int tid = threadIdx.x;
  const int bh  = blockIdx.x, b = bh >> 4, hh = bh & 15;
  const size_t base = ((size_t)b * 64) * DD + (size_t)hh * HDD;

  // Q,K: float4 loads + packed bf16 stores; V: scalar transposed stores
  for (int qd = tid; qd < 1024; qd += 256) {
    int i = qd >> 4, d4 = (qd & 15) * 4;
    float4 vq = *(const float4*)&q[base + (size_t)i * DD + d4];
    float4 vk = *(const float4*)&k[base + (size_t)i * DD + d4];
    unsigned* dq = (unsigned*)&sQ[i][d4];
    dq[0] = pack2bf(vq.x, vq.y); dq[1] = pack2bf(vq.z, vq.w);
    unsigned* dk = (unsigned*)&sK[i][d4];
    dk[0] = pack2bf(vk.x, vk.y); dk[1] = pack2bf(vk.z, vk.w);
    float4 vv = *(const float4*)&v[base + (size_t)i * DD + d4];
    sVT[d4 + 0][i] = f2bf(vv.x);
    sVT[d4 + 1][i] = f2bf(vv.y);
    sVT[d4 + 2][i] = f2bf(vv.z);
    sVT[d4 + 3][i] = f2bf(vv.w);
  }
  __syncthreads();

  const int wave = tid >> 5, lane = tid & 31, half = lane >> 4, l16 = lane & 15;
  const int wm = (wave >> 1) * 16, wn = (wave & 1) * 32;
  v8f acc0 = {}; v8f acc1 = {};
  {
    const unsigned* pA  = (const unsigned*)&sQ[wm + l16][0];
    const unsigned* pB0 = (const unsigned*)&sK[wn + l16][0];
    const unsigned* pB1 = (const unsigned*)&sK[wn + 16 + l16][0];
#pragma unroll
    for (int kc = 0; kc < 64; kc += 32) {
      FragU a, b0, b1; int co = kc >> 1;
#pragma unroll
      for (int j = 0; j < 4; j++) a.u[j] = pA[co + j + 4 * half];
#pragma unroll
      for (int j = 4; j < 8; j++) a.u[j] = pA[co + 8 + (j - 4) + 4 * half];
#pragma unroll
      for (int j = 0; j < 8; j++) { b0.u[j] = pB0[co + (j & 3) + 4 * half + 8 * (j >> 2)]; b1.u[j] = pB1[co + (j & 3) + 4 * half + 8 * (j >> 2)]; }
      acc0 = __builtin_amdgcn_wmma_f32_16x16x32_bf16(false, a.v, false, b0.v, (short)0, acc0, false, false);
      acc1 = __builtin_amdgcn_wmma_f32_16x16x32_bf16(false, a.v, false, b1.v, (short)0, acc1, false, false);
    }
  }
  const float* bb = bias + (size_t)bh * 4096;
#pragma unroll
  for (int r = 0; r < 8; r++) {
    int i = wm + r + 8 * half;
    sS[i][wn + l16]      = acc0[r] * 0.125f + bb[i * 64 + wn + l16];
    sS[i][wn + 16 + l16] = acc1[r] * 0.125f + bb[i * 64 + wn + 16 + l16];
  }
  __syncthreads();
  if (tid < 64) {
    float mx = -1e30f;
    for (int j = 0; j < 64; j++) mx = fmaxf(mx, sS[tid][j]);
    float sum = 0.f;
    for (int j = 0; j < 64; j++) sum += expf(sS[tid][j] - mx);
    float inv = 1.f / sum;
    for (int j = 0; j < 64; j++) sQ[tid][j] = f2bf(expf(sS[tid][j] - mx) * inv);
  }
  __syncthreads();
  v8f acc2 = {}; v8f acc3 = {};
  {
    const unsigned* pA  = (const unsigned*)&sQ[wm + l16][0];
    const unsigned* pB0 = (const unsigned*)&sVT[wn + l16][0];
    const unsigned* pB1 = (const unsigned*)&sVT[wn + 16 + l16][0];
#pragma unroll
    for (int kc = 0; kc < 64; kc += 32) {
      FragU a, b0, b1; int co = kc >> 1;
#pragma unroll
      for (int j = 0; j < 4; j++) a.u[j] = pA[co + j + 4 * half];
#pragma unroll
      for (int j = 4; j < 8; j++) a.u[j] = pA[co + 8 + (j - 4) + 4 * half];
#pragma unroll
      for (int j = 0; j < 8; j++) { b0.u[j] = pB0[co + (j & 3) + 4 * half + 8 * (j >> 2)]; b1.u[j] = pB1[co + (j & 3) + 4 * half + 8 * (j >> 2)]; }
      acc2 = __builtin_amdgcn_wmma_f32_16x16x32_bf16(false, a.v, false, b0.v, (short)0, acc2, false, false);
      acc3 = __builtin_amdgcn_wmma_f32_16x16x32_bf16(false, a.v, false, b1.v, (short)0, acc3, false, false);
    }
  }
#pragma unroll
  for (int r = 0; r < 8; r++) {
    int i = wm + r + 8 * half;
    o[base + (size_t)i * DD + wn + l16]      = acc2[r];
    o[base + (size_t)i * DD + wn + 16 + l16] = acc3[r];
  }
}

// ============================================================
// Policy head: lft = (Pq @ Pk^T)/32 per board (WMMA, K=1024), then
// LC0 73-plane gather of planes 0..63 straight into d_out.
// ============================================================
__global__ void __launch_bounds__(256)
k_policy(const float* __restrict__ pq, const float* __restrict__ pk,
         float* __restrict__ pol)
{
  __shared__ __bf16 sA[64][34];
  __shared__ __bf16 sB[64][34];
  __shared__ float  sS[64][65];
  const int b = blockIdx.x, tid = threadIdx.x;
  const int wave = tid >> 5, lane = tid & 31, half = lane >> 4, l16 = lane & 15;
  const int wm = (wave >> 1) * 16, wn = (wave & 1) * 32;
  const size_t base = (size_t)b * 64 * DD;
  v8f acc0 = {}; v8f acc1 = {};
  for (int kc = 0; kc < DD; kc += 32) {
#pragma unroll
    for (int t = 0; t < 2; t++) {
      int q = tid + t * 256;                 // 512 quads
      int r = q >> 3, c = (q & 7) * 4;
      float4 va = *(const float4*)&pq[base + (size_t)r * DD + kc + c];
      float4 vb = *(const float4*)&pk[base + (size_t)r * DD + kc + c];
      unsigned* da = (unsigned*)&sA[r][c];
      da[0] = pack2bf(va.x, va.y); da[1] = pack2bf(va.z, va.w);
      unsigned* db = (unsigned*)&sB[r][c];
      db[0] = pack2bf(vb.x, vb.y); db[1] = pack2bf(vb.z, vb.w);
    }
    if (kc + 32 < DD) {
      int pr = tid & 63;
      __builtin_prefetch(&pq[base + (size_t)pr * DD + kc + 32], 0, 0);
      __builtin_prefetch(&pk[base + (size_t)pr * DD + kc + 32], 0, 0);
    }
    __syncthreads();
    FragU a, b0, b1;
    const unsigned* pA  = (const unsigned*)&sA[wm + l16][0];
    const unsigned* pB0 = (const unsigned*)&sB[wn + l16][0];
    const unsigned* pB1 = (const unsigned*)&sB[wn + 16 + l16][0];
#pragma unroll
    for (int j = 0; j < 4; j++) a.u[j] = pA[j + 4 * half];
#pragma unroll
    for (int j = 4; j < 8; j++) a.u[j] = pA[8 + (j - 4) + 4 * half];
#pragma unroll
    for (int j = 0; j < 8; j++) { b0.u[j] = pB0[(j & 3) + 4 * half + 8 * (j >> 2)]; b1.u[j] = pB1[(j & 3) + 4 * half + 8 * (j >> 2)]; }
    acc0 = __builtin_amdgcn_wmma_f32_16x16x32_bf16(false, a.v, false, b0.v, (short)0, acc0, false, false);
    acc1 = __builtin_amdgcn_wmma_f32_16x16x32_bf16(false, a.v, false, b1.v, (short)0, acc1, false, false);
    __syncthreads();
  }
#pragma unroll
  for (int r = 0; r < 8; r++) {
    int i = wm + r + 8 * half;
    sS[i][wn + l16]      = acc0[r] * 0.03125f;
    sS[i][wn + 16 + l16] = acc1[r] * 0.03125f;
  }
  __syncthreads();
  for (int idx = tid; idx < 4096; idx += 256) {
    int f = idx >> 6, p = idx & 63;
    int rr = f >> 3, cc = f & 7, nr, nc;
    if (p < 56) { int dd = p / 7, dist = (p % 7) + 1; nr = rr + c_dr[dd] * dist; nc = cc + c_dc[dd] * dist; }
    else        { nr = rr + c_knr[p - 56]; nc = cc + c_knc[p - 56]; }
    float val = NEGV;
    if (nr >= 0 && nr < 8 && nc >= 0 && nc < 8) val = sS[f][nr * 8 + nc];
    pol[(size_t)b * 4672 + f * 73 + p] = val;
  }
}

// ============================================================
// Small VALU kernels (heads, norms, permutes)
// ============================================================
__global__ void k_permx(const float* __restrict__ x, float* __restrict__ xT) {
  int idx = blockIdx.x * blockDim.x + threadIdx.x;
  if (idx >= BB * CC * 64) return;
  int sq = idx & 63, rem = idx >> 6;
  int c = rem % CC, b = rem / CC;
  xT[((size_t)b * 64 + sq) * CC + c] = x[idx];
}

__global__ void k_addpos(float* __restrict__ h, const float* __restrict__ pos) {
  int idx = blockIdx.x * blockDim.x + threadIdx.x;
  if (idx >= TOK * DD) return;
  int m = idx / DD, n = idx - m * DD;
  h[idx] += pos[(m & 63) * DD + n];
}

__global__ void k_ln(const float* __restrict__ x, const float* __restrict__ w,
                     const float* __restrict__ b, float* __restrict__ out, int N) {
  const int row = blockIdx.x, tid = threadIdx.x;
  const float* xr = x + (size_t)row * N;
  __shared__ float red[256];
  float s = 0.f;
  for (int j = tid; j < N; j += 256) s += xr[j];
  red[tid] = s; __syncthreads();
  for (int st = 128; st > 0; st >>= 1) { if (tid < st) red[tid] += red[tid + st]; __syncthreads(); }
  float mu = red[0] / N; __syncthreads();
  float v2 = 0.f;
  for (int j = tid; j < N; j += 256) { float d0 = xr[j] - mu; v2 += d0 * d0; }
  red[tid] = v2; __syncthreads();
  for (int st = 128; st > 0; st >>= 1) { if (tid < st) red[tid] += red[tid + st]; __syncthreads(); }
  float inv = rsqrtf(red[0] / N + 1e-5f);
  for (int j = tid; j < N; j += 256) out[(size_t)row * N + j] = (xr[j] - mu) * inv * w[j] + b[j];
}

// per-head RMS norm on (TOK, H, 64) view; 4 head-rows per block
__global__ void k_rms(float* __restrict__ x, const float* __restrict__ w) {
  const int tid = threadIdx.x, grp = tid >> 6, c = tid & 63;
  long row = (long)blockIdx.x * 4 + grp;
  long tok = row >> 4; int hh = (int)(row & 15);
  float* p = x + tok * DD + hh * HDD + c;
  float v0 = *p;
  __shared__ float red[256];
  red[tid] = v0 * v0; __syncthreads();
  for (int st = 32; st > 0; st >>= 1) { if (c < st) red[tid] += red[tid + st]; __syncthreads(); }
  float ms = red[grp * 64] / 64.f;
  *p = v0 * rsqrtf(ms + 1e-6f) * w[c];
}

__global__ void k_meanpool(const float* __restrict__ h, float* __restrict__ out) {
  int idx = blockIdx.x * blockDim.x + threadIdx.x;
  if (idx >= BB * DD) return;
  int b = idx / DD, d = idx - b * DD;
  float s = 0.f;
  for (int i = 0; i < 64; i++) s += h[((size_t)b * 64 + i) * DD + d];
  out[idx] = s * (1.f / 64.f);
}

// out[m,n] = act(A[m,:]@W[:,n] + bias[n]); act: 0 none, 1 mish, 2 softplus
__global__ void k_small(const float* __restrict__ A, const float* __restrict__ W,
                        const float* __restrict__ bias, float* __restrict__ out,
                        int M, int K, int N, int act) {
  int idx = blockIdx.x * blockDim.x + threadIdx.x;
  if (idx >= M * N) return;
  int m = idx / N, n = idx - m * N;
  float acc = bias ? bias[n] : 0.f;
  for (int k2 = 0; k2 < K; k2++) acc += A[(size_t)m * K + k2] * W[(size_t)k2 * N + n];
  if (act == 1) acc = mishf(acc);
  else if (act == 2) acc = softplusf(acc);
  out[idx] = acc;
}

// promotion planes 64..72 of policy
__global__ void k_up(const float* __restrict__ h, const float* __restrict__ w,
                     const float* __restrict__ b, float* __restrict__ pol) {
  int idx = blockIdx.x * blockDim.x + threadIdx.x;
  if (idx >= TOK * 9) return;
  int j = idx % 9, bsq = idx / 9;
  int f = bsq & 63, bb = bsq >> 6;
  float val = NEGV;
  if ((f >> 3) == 6) {
    float acc = b[j];
    for (int k2 = 0; k2 < DD; k2++) acc += h[(size_t)bsq * DD + k2] * w[k2 * 9 + j];
    val = acc;
  }
  pol[(size_t)bb * 4672 + f * 73 + 64 + j] = val;
}

// ============================================================
static inline void gemm_launch(hipStream_t s, int epi, const float* A, const float* Wm,
                               const float* bias, const float* res, float* C,
                               int M, int N, int K) {
  dim3 g((N + 63) / 64, (M + 127) / 128), blk(256);
  switch (epi) {
    case 0: k_gemm<0><<<g, blk, 0, s>>>(A, Wm, bias, res, C, M, N, K); break;
    case 1: k_gemm<1><<<g, blk, 0, s>>>(A, Wm, bias, res, C, M, N, K); break;
    case 2: k_gemm<2><<<g, blk, 0, s>>>(A, Wm, bias, res, C, M, N, K); break;
    case 3: k_gemm<3><<<g, blk, 0, s>>>(A, Wm, bias, res, C, M, N, K); break;
    default: k_gemm<4><<<g, blk, 0, s>>>(A, Wm, bias, res, C, M, N, K); break;
  }
}

extern "C" void kernel_launch(void* const* d_in, const int* in_sizes, int n_in,
                              void* d_out, int out_size, void* d_ws, size_t ws_size,
                              hipStream_t stream) {
  (void)in_sizes; (void)n_in; (void)out_size; (void)ws_size;
  const float* x       = (const float*)d_in[0];
  const float* embed_w = (const float*)d_in[1];
  const float* embed_b = (const float*)d_in[2];
  const float* pos_emb = (const float*)d_in[3];
  const float* Wq = (const float*)d_in[4];  const float* bq = (const float*)d_in[5];
  const float* Wk = (const float*)d_in[6];  const float* bk = (const float*)d_in[7];
  const float* Wv = (const float*)d_in[8];  const float* bv = (const float*)d_in[9];
  const float* Wo = (const float*)d_in[10]; const float* bo = (const float*)d_in[11];
  const float* qn_w = (const float*)d_in[12]; const float* kn_w = (const float*)d_in[13];
  const float* ln1_w = (const float*)d_in[14]; const float* ln1_b = (const float*)d_in[15];
  const float* ln2_w = (const float*)d_in[16]; const float* ln2_b = (const float*)d_in[17];
  const float* ff_w1 = (const float*)d_in[18]; const float* ff_b1 = (const float*)d_in[19];
  const float* ff_w2 = (const float*)d_in[20]; const float* ff_b2 = (const float*)d_in[21];
  const float* sg_c_w  = (const float*)d_in[22];
  const float* sg_d1_w = (const float*)d_in[23]; const float* sg_d1_b = (const float*)d_in[24];
  const float* sg_ln1_w = (const float*)d_in[25]; const float* sg_ln1_b = (const float*)d_in[26];
  const float* sg_d2_w = (const float*)d_in[27]; const float* sg_d2_b = (const float*)d_in[28];
  const float* sg_ln2_w = (const float*)d_in[29]; const float* sg_ln2_b = (const float*)d_in[30];
  const float* sg_gen_w = (const float*)d_in[31];
  const float* pq_w = (const float*)d_in[32]; const float* pq_b = (const float*)d_in[33];
  const float* pk_w = (const float*)d_in[34]; const float* pk_b = (const float*)d_in[35];
  const float* up_w = (const float*)d_in[36]; const float* up_b = (const float*)d_in[37];
  const float* vt_w = (const float*)d_in[38]; const float* vt_b = (const float*)d_in[39];
  const float* v1_w = (const float*)d_in[40]; const float* v1_b = (const float*)d_in[41];
  const float* v2_w = (const float*)d_in[42]; const float* v2_b = (const float*)d_in[43];
  const float* sc1_w = (const float*)d_in[44]; const float* sc1_b = (const float*)d_in[45];
  const float* sc2_w = (const float*)d_in[46]; const float* sc2_b = (const float*)d_in[47];
  const float* vo1_w = (const float*)d_in[48]; const float* vo1_b = (const float*)d_in[49];
  const float* vo2_w = (const float*)d_in[50]; const float* vo2_b = (const float*)d_in[51];

  // ---- workspace layout (floats) ----
  float* Wp = (float*)d_ws;
  const size_t SZ = (size_t)TOK * DD;          // 8.39M floats
  float* h    = Wp;            Wp += SZ;
  float* tmp  = Wp;            Wp += SZ;       // also used as xT at start
  float* qb   = Wp;            Wp += SZ;
  float* kb   = Wp;            Wp += SZ;
  float* vb   = Wp;            Wp += SZ;       // reused as vt after layers
  float* ao   = Wp;            Wp += SZ;
  float* bbuf = Wp;            Wp += SZ;       // smolgen attention bias (B*H,4096)
  float* ffb  = Wp;            Wp += (size_t)TOK * FFF;
  float* s1   = Wp;            Wp += (size_t)BB * 64 * HCC;   // (B, 2048)
  float* s2   = Wp;            Wp += (size_t)BB * SGHH;       // (B, 256)
  float* s3   = Wp;            Wp += (size_t)BB * HH * GENN;  // (B, 4096)
  float* pooled = Wp;          Wp += (size_t)BB * DD;
  float* v1o  = Wp;            Wp += (size_t)BB * 128;
  float* sct  = Wp;            Wp += (size_t)BB * 32;
  float* vot  = Wp;            Wp += (size_t)BB * 64;

  float* pol        = (float*)d_out;
  float* out_value  = pol + (size_t)BB * 4672;
  float* out_scalar = out_value + (size_t)BB * 3;
  float* out_vol    = out_scalar + (size_t)BB;

  // ---- embed: xT = permute(x); h = xT@embed_w + embed_b + pos ----
  k_permx<<<(BB * CC * 64 + 255) / 256, 256, 0, stream>>>(x, tmp);
  gemm_launch(stream, 1, tmp, embed_w, embed_b, nullptr, h, TOK, DD, CC);
  k_addpos<<<(TOK * DD) / 256, 256, 0, stream>>>(h, pos_emb);

  for (int i = 0; i < LL; i++) {
    // smolgen
    gemm_launch(stream, 0, h, sg_c_w + (size_t)i * DD * HCC, nullptr, nullptr, s1, TOK, HCC, DD);
    gemm_launch(stream, 3, s1, sg_d1_w + (size_t)i * 2048 * SGHH, sg_d1_b + (size_t)i * SGHH,
                nullptr, s2, BB, SGHH, 2048);
    k_ln<<<BB, 256, 0, stream>>>(s2, sg_ln1_w + (size_t)i * SGHH, sg_ln1_b + (size_t)i * SGHH, s2, SGHH);
    gemm_launch(stream, 3, s2, sg_d2_w + (size_t)i * SGHH * 4096, sg_d2_b + (size_t)i * 4096,
                nullptr, s3, BB, 4096, SGHH);
    k_ln<<<BB, 256, 0, stream>>>(s3, sg_ln2_w + (size_t)i * 4096, sg_ln2_b + (size_t)i * 4096, s3, 4096);
    gemm_launch(stream, 0, s3, sg_gen_w + (size_t)i * GENN * 4096, nullptr, nullptr,
                bbuf, BB * HH, 4096, GENN);
    // qkv + rms
    gemm_launch(stream, 1, h, Wq + (size_t)i * DD * DD, bq + (size_t)i * DD, nullptr, qb, TOK, DD, DD);
    gemm_launch(stream, 1, h, Wk + (size_t)i * DD * DD, bk + (size_t)i * DD, nullptr, kb, TOK, DD, DD);
    gemm_launch(stream, 1, h, Wv + (size_t)i * DD * DD, bv + (size_t)i * DD, nullptr, vb, TOK, DD, DD);
    k_rms<<<(TOK * HH) / 4, 256, 0, stream>>>(qb, qn_w + (size_t)i * HDD);
    k_rms<<<(TOK * HH) / 4, 256, 0, stream>>>(kb, kn_w + (size_t)i * HDD);
    // attention
    k_attn<<<BB * HH, 256, 0, stream>>>(qb, kb, vb, bbuf, ao);
    // out proj + residual + LN
    gemm_launch(stream, 4, ao, Wo + (size_t)i * DD * DD, bo + (size_t)i * DD, h, tmp, TOK, DD, DD);
    k_ln<<<TOK, 256, 0, stream>>>(tmp, ln1_w + (size_t)i * DD, ln1_b + (size_t)i * DD, h, DD);
    // FF
    gemm_launch(stream, 2, h, ff_w1 + (size_t)i * DD * FFF, ff_b1 + (size_t)i * FFF, nullptr, ffb, TOK, FFF, DD);
    gemm_launch(stream, 4, ffb, ff_w2 + (size_t)i * FFF * DD, ff_b2 + (size_t)i * DD, h, tmp, TOK, DD, FFF);
    k_ln<<<TOK, 256, 0, stream>>>(tmp, ln2_w + (size_t)i * DD, ln2_b + (size_t)i * DD, h, DD);
  }

  // ---- policy head ----
  gemm_launch(stream, 1, h, pq_w, pq_b, nullptr, qb, TOK, DD, DD);
  gemm_launch(stream, 1, h, pk_w, pk_b, nullptr, kb, TOK, DD, DD);
  k_policy<<<BB, 256, 0, stream>>>(qb, kb, pol);
  k_up<<<(TOK * 9 + 255) / 256, 256, 0, stream>>>(h, up_w, up_b, pol);

  // ---- value head ----
  gemm_launch(stream, 2, h, vt_w, vt_b, nullptr, vb, TOK, 128, DD);          // vt (B,64*128)
  gemm_launch(stream, 2, vb, v1_w, v1_b, nullptr, v1o, BB, 128, 64 * 128);   // (128,128)
  k_small<<<(BB * 3 + 255) / 256, 256, 0, stream>>>(v1o, v2_w, v2_b, out_value, BB, 128, 3, 0);

  // ---- scalar + vol heads ----
  k_meanpool<<<(BB * DD) / 256, 256, 0, stream>>>(h, pooled);
  k_small<<<(BB * 32 + 255) / 256, 256, 0, stream>>>(pooled, sc1_w, sc1_b, sct, BB, DD, 32, 1);
  k_small<<<1, 256, 0, stream>>>(sct, sc2_w, sc2_b, out_scalar, BB, 32, 1, 0);
  k_small<<<(BB * 64 + 255) / 256, 256, 0, stream>>>(pooled, vo1_w, vo1_b, vot, BB, DD, 64, 1);
  k_small<<<(BB * 3 + 255) / 256, 256, 0, stream>>>(vot, vo2_w, vo2_b, out_vol, BB, 64, 3, 2);
}
